// AutoregressiveLoop_25838523253320
// MI455X (gfx1250) — compile-verified, matrix operand order checked by reading, not run on hardware
//
#include <hip/hip_runtime.h>
#include <hip/hip_bf16.h>
#include <stddef.h>

// ---------------------------------------------------------------------------
// Autoregressive 2-layer LSTM (H=2048, batch=1, 256 steps) for MI455X gfx1250.
//
// Strategy:
//  * prep kernels: convert W_hh0 / W_ih1 / W_hh1 (8192x2048 fp32) -> bf16,
//    pre-tiled in the exact WMMA A-matrix (16x32, 16-bit) VGPR layout so the
//    hot loop issues contiguous b128 loads. ~96 MB total -> L2-resident.
//  * one persistent kernel (64 blocks x 256 threads = 512 waves) runs all
//    steps; grid-wide sense-reversing barriers separate the phases.
//  * each wave computes one 16-row gate tile with v_wmma_f32_16x16x32_bf16,
//    B = h broadcast across the 16 columns (built from LDS-staged bf16 h).
// ---------------------------------------------------------------------------

#define NJ 7
#define H 2048
#define G4H 8192              // 4*H
#define ROWTILES 512          // 8192/16
#define KTILES 64             // 2048/32
#define NBLOCKS 64
#define NTHREADS 256
#define WAVES_PER_BLOCK 8

typedef __attribute__((ext_vector_type(16))) __bf16 v16bf;
typedef __attribute__((ext_vector_type(8)))  float  v8f;

union BF16x16 {
    v16bf v;
    uint4 q[2];
};

__device__ __forceinline__ unsigned short f2bf(float f) {
    unsigned u = __float_as_uint(f);
    unsigned r = u + 0x7fffu + ((u >> 16) & 1u);   // round-to-nearest-even
    return (unsigned short)(r >> 16);
}

__device__ __forceinline__ float sigmoidf(float z) {
    return 1.0f / (1.0f + __expf(-z));
}

__device__ __forceinline__ v8f wmma_bf16(v16bf a, v16bf b, v8f c) {
    return __builtin_amdgcn_wmma_f32_16x16x32_bf16(
        /*neg_a=*/false, a, /*neg_b=*/false, b,
        /*c_mod=*/(short)0, c, /*reuse_a=*/false, /*reuse_b=*/false);
}

// sense-reversing grid barrier (bar[0]=counter, bar[1]=generation)
__device__ __forceinline__ void grid_barrier(unsigned* bar) {
    __syncthreads();
    if (threadIdx.x == 0) {
        __threadfence();                       // release
        volatile unsigned* genp = bar + 1;
        unsigned gen = *genp;
        unsigned arrived = atomicAdd(bar, 1u);
        if (arrived == (unsigned)gridDim.x - 1u) {
            bar[0] = 0u;
            __threadfence();
            atomicAdd(bar + 1, 1u);
        } else {
            while (*genp == gen) { __builtin_amdgcn_s_sleep(1); }
        }
        __threadfence();                       // acquire
    }
    __syncthreads();
}

// ---------------------------------------------------------------------------
// prep: fp32 (8192x2048) -> bf16 tiled into WMMA 16-bit A-matrix layout.
// tile (rt,kt) = 512 ushorts; within tile: lane*16 + half.
//   lane = m + 16*((k&15)>=8) ; half = (k&7) + 8*((k&31)>=16)
// ---------------------------------------------------------------------------
__global__ __launch_bounds__(256) void prep_weights(const float* __restrict__ src,
                                                    unsigned short* __restrict__ dst,
                                                    int rows, int cols) {
    size_t e = (size_t)blockIdx.x * blockDim.x + threadIdx.x;
    size_t total = (size_t)rows * cols;
    if (e >= total) return;
    int r = (int)(e / cols);
    int c = (int)(e % cols);
    int ktiles = cols >> 5;
    int tile = (r >> 4) * ktiles + (c >> 5);
    int lane = (r & 15) + (((c & 15) >= 8) ? 16 : 0);
    int half = (c & 7) + (((c & 31) >= 16) ? 8 : 0);
    dst[(size_t)tile * 512 + lane * 16 + half] = f2bf(src[e]);
}

__global__ __launch_bounds__(256) void prep_state(const float* __restrict__ q,
                                                  const float* __restrict__ qd,
                                                  const float* __restrict__ hh,
                                                  const float* __restrict__ hc,
                                                  const float* __restrict__ delay,
                                                  float* x, float* h0, float* c0,
                                                  float* h1, float* c1, unsigned* bar) {
    int i = blockIdx.x * 256 + threadIdx.x;
    if (i < H) {
        h0[i] = hh[i];
        h1[i] = hh[H + i];
        c0[i] = hc[i];
        c1[i] = hc[H + i];
    }
    if (blockIdx.x == 0) {
        if (threadIdx.x < NJ) {
            x[threadIdx.x]      = q[threadIdx.x];
            x[NJ + threadIdx.x] = qd[threadIdx.x];
        }
        if (threadIdx.x == NJ) { x[14] = delay[0]; x[15] = 0.0f; }
        if (threadIdx.x == NJ + 1) { bar[0] = 0u; bar[1] = 0u; }
    }
}

// ---------------------------------------------------------------------------
// persistent main kernel
// ---------------------------------------------------------------------------
__global__ __launch_bounds__(NTHREADS) void lstm_persistent(
    const unsigned short* __restrict__ Whh0b,
    const unsigned short* __restrict__ Wih1b,
    const unsigned short* __restrict__ Whh1b,
    const float* __restrict__ Wih0,   // 8192 x 15 fp32
    const float* __restrict__ bih0, const float* __restrict__ bhh0,
    const float* __restrict__ bih1, const float* __restrict__ bhh1,
    const float* __restrict__ fcW,    // 14 x 2048 fp32
    const float* __restrict__ fcb,    // 14
    float* __restrict__ x,            // [16]: q(7), qd(7), delay, pad
    float* __restrict__ h0, float* __restrict__ c0,
    float* __restrict__ h1, float* __restrict__ c1,
    float* __restrict__ gates,        // [8192]
    unsigned* __restrict__ bar,
    const int* __restrict__ steps_p,
    float* __restrict__ out)
{
    const int tid  = threadIdx.x;
    const int lane = tid & 31;
    const int wave = tid >> 5;
    const int gw   = blockIdx.x * WAVES_PER_BLOCK + wave;   // 0..511 == row tile
    const int grp  = lane >> 4;                             // B-operand K-group

    __shared__ alignas(16) unsigned short sh0[H];
    __shared__ alignas(16) unsigned short sh1[H];
    __shared__ float sx[16];
    __shared__ float sextra[WAVES_PER_BLOCK * 16];
    __shared__ float sred[NTHREADS];
    __shared__ float sh1f[H];

    const int nsteps = *steps_p;

    for (int s = 0; s < nsteps; ++s) {
        // ---- stage x and h0 (bf16) in LDS ----------------------------------
        if (tid < 16) sx[tid] = x[tid];
        for (int i = tid; i < H; i += NTHREADS) sh0[i] = f2bf(h0[i]);
        __syncthreads();

        // ---- phase A: gates = Whh0@h0 + Wih0@x + (bih0+bhh0) --------------
        {
            v8f acc = {0.f,0.f,0.f,0.f,0.f,0.f,0.f,0.f};
            const uint4* abase =
                (const uint4*)(Whh0b + (size_t)gw * KTILES * 512) + lane * 2;
            #pragma unroll 4
            for (int kt = 0; kt < KTILES; ++kt) {
                BF16x16 A, B;
                A.q[0] = abase[0];
                A.q[1] = abase[1];
                abase += 64;   // 512 ushorts = 64 uint4 per tile
                B.q[0] = *(const uint4*)&sh0[kt * 32 + grp * 8];
                B.q[1] = *(const uint4*)&sh0[kt * 32 + 16 + grp * 8];
                acc = wmma_bf16(A.v, B.v, acc);
            }
            if (lane < 16) {
                int r = gw * 16 + lane;
                float e = bih0[r] + bhh0[r];
                const float* wrow = Wih0 + (size_t)r * 15;
                #pragma unroll
                for (int k = 0; k < 15; ++k) e += wrow[k] * sx[k];
                sextra[wave * 16 + lane] = e;
            }
            // D layout: lane0 holds rows 0..7 (col 0), lane16 rows 8..15
            if (lane == 0) {
                #pragma unroll
                for (int r = 0; r < 8; ++r)
                    gates[gw * 16 + r] = acc[r] + sextra[wave * 16 + r];
            } else if (lane == 16) {
                #pragma unroll
                for (int r = 0; r < 8; ++r)
                    gates[gw * 16 + 8 + r] = acc[r] + sextra[wave * 16 + 8 + r];
            }
        }
        grid_barrier(bar);

        // ---- phase B: h0,c0 cell update (blocks 0..7 cover 2048 elems) ----
        if (blockIdx.x < 8) {
            int j = blockIdx.x * NTHREADS + tid;
            float gi = gates[j], gf = gates[H + j];
            float gg = gates[2 * H + j], go = gates[3 * H + j];
            float cn = sigmoidf(gf) * c0[j] + sigmoidf(gi) * tanhf(gg);
            c0[j] = cn;
            h0[j] = sigmoidf(go) * tanhf(cn);
        }
        grid_barrier(bar);

        // ---- stage new h0 and h1 (bf16) -----------------------------------
        for (int i = tid; i < H; i += NTHREADS) {
            sh0[i] = f2bf(h0[i]);
            sh1[i] = f2bf(h1[i]);
        }
        __syncthreads();

        // ---- phase C: gates = Wih1@h0 + Whh1@h1 + (bih1+bhh1) -------------
        {
            v8f acc = {0.f,0.f,0.f,0.f,0.f,0.f,0.f,0.f};
            const uint4* a0 =
                (const uint4*)(Wih1b + (size_t)gw * KTILES * 512) + lane * 2;
            const uint4* a1 =
                (const uint4*)(Whh1b + (size_t)gw * KTILES * 512) + lane * 2;
            #pragma unroll 4
            for (int kt = 0; kt < KTILES; ++kt) {
                BF16x16 A, B;
                A.q[0] = a0[0]; A.q[1] = a0[1]; a0 += 64;
                B.q[0] = *(const uint4*)&sh0[kt * 32 + grp * 8];
                B.q[1] = *(const uint4*)&sh0[kt * 32 + 16 + grp * 8];
                acc = wmma_bf16(A.v, B.v, acc);
            }
            #pragma unroll 4
            for (int kt = 0; kt < KTILES; ++kt) {
                BF16x16 A, B;
                A.q[0] = a1[0]; A.q[1] = a1[1]; a1 += 64;
                B.q[0] = *(const uint4*)&sh1[kt * 32 + grp * 8];
                B.q[1] = *(const uint4*)&sh1[kt * 32 + 16 + grp * 8];
                acc = wmma_bf16(A.v, B.v, acc);
            }
            if (lane < 16) {
                int r = gw * 16 + lane;
                sextra[wave * 16 + lane] = bih1[r] + bhh1[r];
            }
            if (lane == 0) {
                #pragma unroll
                for (int r = 0; r < 8; ++r)
                    gates[gw * 16 + r] = acc[r] + sextra[wave * 16 + r];
            } else if (lane == 16) {
                #pragma unroll
                for (int r = 0; r < 8; ++r)
                    gates[gw * 16 + 8 + r] = acc[r] + sextra[wave * 16 + 8 + r];
            }
        }
        grid_barrier(bar);

        // ---- phase D+E: block 0 does h1/c1 update, fc head, state update --
        if (blockIdx.x == 0) {
            for (int j = tid; j < H; j += NTHREADS) {
                float gi = gates[j], gf = gates[H + j];
                float gg = gates[2 * H + j], go = gates[3 * H + j];
                float cn = sigmoidf(gf) * c1[j] + sigmoidf(gi) * tanhf(gg);
                c1[j] = cn;
                float hn = sigmoidf(go) * tanhf(cn);
                h1[j] = hn;
                sh1f[j] = hn;
            }
            __syncthreads();
            float part = 0.0f;
            if (tid < 2 * NJ * 16) {
                int j  = tid >> 4;
                int s2 = tid & 15;
                const float* wrow = fcW + (size_t)j * H;
                for (int k = s2; k < H; k += 16) part += wrow[k] * sh1f[k];
            }
            sred[tid] = part;
            __syncthreads();
            if (tid < 2 * NJ) {
                float t = 0.0f;
                #pragma unroll
                for (int u = 0; u < 16; ++u) t += sred[tid * 16 + u];
                float res = (t + fcb[tid]) * 0.1f;           // SCALE
                res = fminf(0.2f, fmaxf(-0.2f, res));        // clip
                x[tid] += res;                               // q / qd update
            }
            if (tid == 2 * NJ) x[14] += 0.01f;               // delay += DT_NORM
        }
        grid_barrier(bar);
    }

    // ---- outputs: [q(7), qd(7), h0, h1, c0, c1] ---------------------------
    if (blockIdx.x == 0) {
        if (tid < 14) out[tid] = x[tid];
        for (int i = tid; i < H; i += NTHREADS) {
            out[14 + i]         = h0[i];
            out[14 + H + i]     = h1[i];
            out[14 + 2 * H + i] = c0[i];
            out[14 + 3 * H + i] = c1[i];
        }
    }
}

// ---------------------------------------------------------------------------
extern "C" void kernel_launch(void* const* d_in, const int* in_sizes, int n_in,
                              void* d_out, int out_size, void* d_ws, size_t ws_size,
                              hipStream_t stream) {
    (void)in_sizes; (void)n_in; (void)out_size; (void)ws_size;

    const float* q      = (const float*)d_in[0];
    const float* qd     = (const float*)d_in[1];
    const float* hh     = (const float*)d_in[2];
    const float* hc     = (const float*)d_in[3];
    const float* delay  = (const float*)d_in[4];
    const float* Wih0   = (const float*)d_in[5];
    const float* Whh0   = (const float*)d_in[6];
    const float* bih0   = (const float*)d_in[7];
    const float* bhh0   = (const float*)d_in[8];
    const float* Wih1   = (const float*)d_in[9];
    const float* Whh1   = (const float*)d_in[10];
    const float* bih1   = (const float*)d_in[11];
    const float* bhh1   = (const float*)d_in[12];
    const float* fcW    = (const float*)d_in[13];
    const float* fcb    = (const float*)d_in[14];
    const int*   steps  = (const int*)d_in[15];

    // workspace layout: 3 tiled bf16 matrices (32 MiB each) + state + barrier
    const size_t WELEMS = (size_t)G4H * H;             // 16,777,216
    unsigned short* Whh0b = (unsigned short*)d_ws;
    unsigned short* Wih1b = Whh0b + WELEMS;
    unsigned short* Whh1b = Wih1b + WELEMS;
    float* state = (float*)(Whh1b + WELEMS);
    float* x     = state;          // 16
    float* h0    = state + 16;     // 2048
    float* c0    = h0 + H;
    float* h1    = c0 + H;
    float* c1    = h1 + H;
    float* gates = c1 + H;         // 8192
    unsigned* bar = (unsigned*)(gates + G4H);

    const int cvtBlocks = (int)((WELEMS + 255) / 256);
    prep_weights<<<cvtBlocks, 256, 0, stream>>>(Whh0, Whh0b, G4H, H);
    prep_weights<<<cvtBlocks, 256, 0, stream>>>(Wih1, Wih1b, G4H, H);
    prep_weights<<<cvtBlocks, 256, 0, stream>>>(Whh1, Whh1b, G4H, H);
    prep_state<<<8, 256, 0, stream>>>(q, qd, hh, hc, delay, x, h0, c0, h1, c1, bar);

    lstm_persistent<<<NBLOCKS, NTHREADS, 0, stream>>>(
        Whh0b, Wih1b, Whh1b, Wih0, bih0, bhh0, bih1, bhh1, fcW, fcb,
        x, h0, c0, h1, c1, gates, bar, steps, (float*)d_out);
}
